// QLinearExperimentalPyTiling_59425167507994
// MI455X (gfx1250) — 8-bit matrix operand order repaired and checked by reading, compiled, not run on hardware
//
#include <hip/hip_runtime.h>
#include <cstdint>

#define M_DIM 4096
#define N_DIM 4096
#define K_DIM 4096

#define X_SCALE 0.03f
#define Y_SCALE 0.008f

#define BLK_M 256
#define BLK_N 128
#define KSTEP 64
#define LDS_STRIDE 80

typedef __attribute__((ext_vector_type(8))) int   v8i;
typedef __attribute__((ext_vector_type(4))) float v4f;

__device__ __forceinline__ int clamp_i8(int q) {
    return q < -128 ? -128 : (q > 127 ? 127 : q);
}

__global__ void __launch_bounds__(256) quantize_f32_to_i8(
    const float* __restrict__ in, uint32_t* __restrict__ out,
    float inv_scale, int n4) {
    int idx = blockIdx.x * blockDim.x + threadIdx.x;
    if (idx >= n4) return;
    v4f v = __builtin_nontemporal_load(reinterpret_cast<const v4f*>(in) + idx);
    int q0 = clamp_i8(__float2int_rn(v.x * inv_scale));
    int q1 = clamp_i8(__float2int_rn(v.y * inv_scale));
    int q2 = clamp_i8(__float2int_rn(v.z * inv_scale));
    int q3 = clamp_i8(__float2int_rn(v.w * inv_scale));
    uint32_t pack = (uint32_t)(q0 & 0xff)
                  | ((uint32_t)(q1 & 0xff) << 8)
                  | ((uint32_t)(q2 & 0xff) << 16)
                  | ((uint32_t)(q3 & 0xff) << 24);
    out[idx] = pack;
}

__global__ void __launch_bounds__(256) gemm_i8_wmma(
    const int8_t* __restrict__ xq, const int8_t* __restrict__ wq,
    float* __restrict__ out, float deq) {
    __shared__ int8_t sA[2][BLK_M * LDS_STRIDE];
    __shared__ int8_t sB[2][BLK_N * LDS_STRIDE];

    const int tid  = threadIdx.x;
    const int lane = tid & 31;
    const int wave = tid >> 5;
    const int wr   = wave >> 1;
    const int wc   = wave & 1;
    const int half = lane >> 4;
    const int l16  = lane & 15;

    const int blockM = blockIdx.y * BLK_M;
    const int blockN = blockIdx.x * BLK_N;

    auto stage = [&](int kofs, int buf) {
#pragma unroll
        for (int p = 0; p < 4; ++p) {
            int c = p * 256 + tid;
            int row = c >> 2, kc = c & 3;
            const int8_t* gp = xq + (size_t)(blockM + row) * K_DIM + kofs + kc * 16;
            uint32_t lo = (uint32_t)(uintptr_t)(&sA[buf][row * LDS_STRIDE + kc * 16]);
            asm volatile("global_load_async_to_lds_b128 %0, %1, off"
                         :: "v"(lo), "v"(gp) : "memory");
        }
#pragma unroll
        for (int p = 0; p < 2; ++p) {
            int c = p * 256 + tid;
            int row = c >> 2, kc = c & 3;
            const int8_t* gp = wq + (size_t)(blockN + row) * K_DIM + kofs + kc * 16;
            uint32_t lo = (uint32_t)(uintptr_t)(&sB[buf][row * LDS_STRIDE + kc * 16]);
            asm volatile("global_load_async_to_lds_b128 %0, %1, off"
                         :: "v"(lo), "v"(gp) : "memory");
        }
    };

    v8i acc[4][4];
#pragma unroll
    for (int i = 0; i < 4; ++i)
#pragma unroll
        for (int j = 0; j < 4; ++j)
            acc[i][j] = (v8i){0, 0, 0, 0, 0, 0, 0, 0};

    stage(0, 0);
    asm volatile("s_wait_asynccnt 0x0" ::: "memory");
    __syncthreads();

    int cur = 0;
    for (int ks = 0; ks < K_DIM; ks += KSTEP) {
        if (ks + KSTEP < K_DIM) stage(ks + KSTEP, cur ^ 1);

        v8i afrag[4];
#pragma unroll
        for (int mt = 0; mt < 4; ++mt) {
            const int8_t* ap =
                &sA[cur][(wr * 64 + mt * 16 + l16) * LDS_STRIDE + half * 8];
            uint2 p0 = *reinterpret_cast<const uint2*>(ap);
            uint2 p1 = *reinterpret_cast<const uint2*>(ap + 16);
            uint2 p2 = *reinterpret_cast<const uint2*>(ap + 32);
            uint2 p3 = *reinterpret_cast<const uint2*>(ap + 48);
            v8i a;
            a[0] = (int)p0.x; a[1] = (int)p0.y;
            a[2] = (int)p1.x; a[3] = (int)p1.y;
            a[4] = (int)p2.x; a[5] = (int)p2.y;
            a[6] = (int)p3.x; a[7] = (int)p3.y;
            afrag[mt] = a;
        }
        v8i bfrag[4];
#pragma unroll
        for (int nt = 0; nt < 4; ++nt) {
            const int8_t* bp =
                &sB[cur][(wc * 64 + nt * 16 + l16) * LDS_STRIDE + half * 8];
            uint2 q0 = *reinterpret_cast<const uint2*>(bp);
            uint2 q1 = *reinterpret_cast<const uint2*>(bp + 16);
            uint2 q2 = *reinterpret_cast<const uint2*>(bp + 32);
            uint2 q3 = *reinterpret_cast<const uint2*>(bp + 48);
            v8i b;
            b[0] = (int)q0.x; b[1] = (int)q0.y; b[2] = (int)q1.x; b[3] = (int)q1.y;
            b[4] = (int)q2.x; b[5] = (int)q2.y; b[6] = (int)q3.x; b[7] = (int)q3.y;
            bfrag[nt] = b;
        }
#pragma unroll
        for (int mt = 0; mt < 4; ++mt)
#pragma unroll
            for (int nt = 0; nt < 4; ++nt)
                acc[mt][nt] = __builtin_amdgcn_wmma_i32_16x16x64_iu8(
                     true, afrag[mt],
                     true, bfrag[nt],
                    acc[mt][nt],
                     false,  false);

        asm volatile("s_wait_asynccnt 0x0" ::: "memory");
        __syncthreads();
        cur ^= 1;
    }

#pragma unroll
    for (int mt = 0; mt < 4; ++mt) {
#pragma unroll
        for (int nt = 0; nt < 4; ++nt) {
            const int row0 = blockM + wr * 64 + mt * 16 + half * 8;
            const int col  = blockN + wc * 64 + nt * 16 + l16;
#pragma unroll
            for (int r = 0; r < 8; ++r) {
                __builtin_nontemporal_store(
                    (float)acc[mt][nt][r] * deq,
                    out + (size_t)(row0 + r) * N_DIM + col);
            }
        }
    }
}

extern "C" void kernel_launch(void* const* d_in, const int* in_sizes, int n_in,
                              void* d_out, int out_size, void* d_ws, size_t ws_size,
                              hipStream_t stream) {
    const float* x = (const float*)d_in[0];
    const float* w = (const float*)d_in[1];
    float* out = (float*)d_out;

    int8_t* xq = (int8_t*)d_ws;
    int8_t* wq = xq + (size_t)M_DIM * K_DIM;

    const int n4 = (M_DIM * K_DIM) / 4;
    const int qblocks = n4 / 256;

    quantize_f32_to_i8<<<qblocks, 256, 0, stream>>>(
        x, (uint32_t*)xq, 1.0f / X_SCALE, n4);
    quantize_f32_to_i8<<<qblocks, 256, 0, stream>>>(
        w, (uint32_t*)wq, 1.0f / Y_SCALE, n4);

    dim3 grid(N_DIM / BLK_N, M_DIM / BLK_M);
    gemm_i8_wmma<<<grid, 256, 0, stream>>>(
        xq, wq, out, X_SCALE * Y_SCALE);
}
